// DownTransitionMax_19868518711859
// MI455X (gfx1250) — matrix operand order repaired and checked by reading, not compiled, not run on hardware
//
#include <hip/hip_runtime.h>
#include <hip/hip_bf16.h>

// ---------------------------------------------------------------------------
// CDNA5 (gfx1250) sparse DownTransition block.
//   h    = ELU(BN(feats @ Wd))                    (1x1 conv, recomputed twice)
//   down = segment_max(h, pool_seg)               (uint-encoded atomic max)
//   o    = ELU(BN(sparse_conv27(o, Wc)))  x2      (gather + WMMA bf16, fp32 acc)
//   out  = ELU(o + down)
// Conv weights are staged as WMMA B-fragments in the 320KB WGP LDS (216KB),
// inner loop is batched loads -> WMMA bursts (v_wmma_f32_16x16x32_bf16).
// ---------------------------------------------------------------------------

typedef __attribute__((ext_vector_type(16))) __bf16 v16bf;
typedef __attribute__((ext_vector_type(8)))  __bf16 v8bf;
typedef __attribute__((ext_vector_type(8)))  float  v8f;

#define BN_EPS 1e-5f
#define KV 27

__device__ __forceinline__ float elu1(float x) {
    return x > 0.f ? x : expm1f(x);
}

// order-preserving float <-> uint encoding for atomic max pooling
__device__ __forceinline__ unsigned fenc(float f) {
    unsigned u = __float_as_uint(f);
    return (u & 0x80000000u) ? ~u : (u | 0x80000000u);
}
__device__ __forceinline__ float fdec(unsigned u) {
    unsigned b = (u & 0x80000000u) ? (u & 0x7FFFFFFFu) : ~u;
    return __uint_as_float(b);
}

__device__ __forceinline__ v16bf ld16(const __bf16* p0, const __bf16* p1) {
    v8bf lo = *(const v8bf*)p0;
    v8bf hi = *(const v8bf*)p1;
    return __builtin_shufflevector(lo, hi, 0,1,2,3,4,5,6,7,8,9,10,11,12,13,14,15);
}

#define WMMA_BF16(A, B, C) \
    __builtin_amdgcn_wmma_f32_16x16x32_bf16(false, (A), false, (B), (short)0, (C), false, false)

// ---------------------------------------------------------------------------
// generic fill (stats zero, pool init, idx-table default, pad row)
// ---------------------------------------------------------------------------
__global__ void k_fill_u32(unsigned* p, long long n, unsigned v) {
    long long i = (long long)blockIdx.x * blockDim.x + threadIdx.x;
    if (i < n) p[i] = v;
}

// ---------------------------------------------------------------------------
// Repack fp32 weights [kvol][64][64] (or Wd [32][64] with kvol=1,ksteps=1)
// into WMMA B-fragment layout (bf16):
//   frag f = (k*ksteps+ks)*4 + t ; elem (lane,e) = W[k][32*ks + 8*(lane/16) + (e&7) + 16*(e>>3)][16*t + lane%16]
// ---------------------------------------------------------------------------
__global__ void k_build_bfrag(const float* __restrict__ W, __bf16* __restrict__ frag,
                              int kvol, int ksteps) {
    int i = blockIdx.x * blockDim.x + threadIdx.x;
    int total = kvol * ksteps * 4 * 512;
    if (i >= total) return;
    int e    = i & 15;
    int lane = (i >> 4) & 31;
    int t    = (i >> 9) & 3;
    int rest = i >> 11;               // k*ksteps + ks
    int ks   = rest % ksteps;
    int k    = rest / ksteps;
    int K = 32 * ks + 8 * (lane >> 4) + (e & 7) + ((e >> 3) << 4);
    int Ncol = 16 * t + (lane & 15);
    frag[i] = (__bf16)W[(k * 64 + K) * 64 + Ncol];
}

// ---------------------------------------------------------------------------
// Invert scatter pair-lists into dense gather table: tab[k][out_row] = in_row
// (default M = zero row, pre-filled by k_fill_u32)
// ---------------------------------------------------------------------------
__global__ void k_build_idx(const int* __restrict__ ci, const int* __restrict__ co,
                            int* __restrict__ tab, int L, int M) {
    long long i = (long long)blockIdx.x * blockDim.x + threadIdx.x;
    if (i < (long long)KV * L) {
        int k = (int)(i / L);
        int o = co[i];                 // pads point at row M (dummy slot)
        tab[(long long)k * (M + 1) + o] = ci[i];
    }
}

// ---------------------------------------------------------------------------
// 1x1 down-conv: feats[N,32] @ Wd[32,64] via WMMA bf16. B-fragments are loop
// invariant -> loaded into registers once per wave.
// MODE 0: per-channel sum/sumsq (LDS reduce -> global atomics).
// MODE 1: BN+ELU then atomic-max pool into down_u (uint-encoded).
// ---------------------------------------------------------------------------
template <int MODE>
__global__ __launch_bounds__(256)
void k_down_gemm(const float* __restrict__ feats, const __bf16* __restrict__ wdfrag,
                 int N, float* __restrict__ g_sum, float* __restrict__ g_sq,
                 const float* __restrict__ scale, const float* __restrict__ bias,
                 const int* __restrict__ pool_seg, unsigned* __restrict__ down_u) {
    __shared__ float s_sum[64], s_sq[64];
    int tid = threadIdx.x;
    if (MODE == 0) {
        if (tid < 64) { s_sum[tid] = 0.f; s_sq[tid] = 0.f; }
        __syncthreads();
    }
    int lane = tid & 31;
    int wave = tid >> 5;
    long long row0 = ((long long)blockIdx.x * 8 + wave) * 16;
    bool active = row0 < N;

    v8f acc[4] = {};
    if (active) {
        v16bf bfr[4];
#pragma unroll
        for (int t = 0; t < 4; t++)
            bfr[t] = ld16(wdfrag + ((t * 32 + lane) * 16),
                          wdfrag + ((t * 32 + lane) * 16) + 8);
        int r = (int)row0 + (lane & 15);
        bool vrow = r < N;
        const float* arow = feats + (long long)(vrow ? r : 0) * 32;
        int base = (lane < 16) ? 0 : 8;
        v16bf a;
#pragma unroll
        for (int e = 0; e < 16; e++) {
            int K = base + (e & 7) + ((e >> 3) << 4);
            a[e] = (__bf16)(vrow ? arow[K] : 0.f);
        }
#pragma unroll
        for (int t = 0; t < 4; t++)
            acc[t] = WMMA_BF16(a, bfr[t], acc[t]);
    }

    int half = (lane < 16) ? 0 : 8;
    if (MODE == 0) {
        if (active) {
#pragma unroll
            for (int t = 0; t < 4; t++) {
                float s = 0.f, q = 0.f;
#pragma unroll
                for (int rr = 0; rr < 8; rr++) {
                    long long row = row0 + half + rr;
                    if (row < N) { float x = acc[t][rr]; s += x; q += x * x; }
                }
                int c = 16 * t + (lane & 15);
                atomicAdd(&s_sum[c], s);
                atomicAdd(&s_sq[c], q);
            }
        }
        __syncthreads();
        if (tid < 64) {
            atomicAdd(&g_sum[tid], s_sum[tid]);
            atomicAdd(&g_sq[tid], s_sq[tid]);
        }
    } else {
        if (active) {
#pragma unroll
            for (int t = 0; t < 4; t++) {
                int c = 16 * t + (lane & 15);
                float sc = scale[c], bi = bias[c];
#pragma unroll
                for (int rr = 0; rr < 8; rr++) {
                    long long row = row0 + half + rr;
                    if (row < N) {
                        float e = elu1(sc * acc[t][rr] + bi);
                        int seg = pool_seg[row];
                        atomicMax(&down_u[(long long)seg * 64 + c], fenc(e));
                    }
                }
            }
        }
    }
}

// ---------------------------------------------------------------------------
// BN finalize: scale = g*rsqrt(var+eps), bias = b - mean*scale  (biased var)
// ---------------------------------------------------------------------------
__global__ void k_bn_finalize(const float* __restrict__ g, const float* __restrict__ b,
                              const float* __restrict__ sum, const float* __restrict__ sq,
                              float count, float* __restrict__ scale, float* __restrict__ bias) {
    int c = threadIdx.x;
    if (c < 64) {
        float m = sum[c] / count;
        float v = sq[c] / count - m * m;
        float s = g[c] * rsqrtf(v + BN_EPS);
        scale[c] = s;
        bias[c] = b[c] - m * s;
    }
}

// ---------------------------------------------------------------------------
// Decode pooled maxima: uint-encoded -> float bits (in place) + bf16 copy.
// ---------------------------------------------------------------------------
__global__ void k_decode_pool(unsigned* __restrict__ du, __bf16* __restrict__ in_bf,
                              long long n) {
    long long i = (long long)blockIdx.x * blockDim.x + threadIdx.x;
    if (i < n) {
        float f = fdec(du[i]);
        du[i] = __float_as_uint(f);   // becomes fp32 "down" for the residual
        in_bf[i] = (__bf16)f;
    }
}

// ---------------------------------------------------------------------------
// 27-offset sparse conv, gather form: out[m,:] = sum_k in_bf[tab[k][m],:] @ W[k]
// 16 waves/block; all 27 offsets' B-fragments staged in LDS (216KB of the
// 320KB WGP LDS); each wave owns TWO 16-row tiles so every LDS B-fragment
// feeds 2 WMMAs. Inner body: batched (global A + LDS B) loads, then a
// 16-WMMA burst per offset; full unroll lets the scheduler overlap offsets.
// ---------------------------------------------------------------------------
__global__ __launch_bounds__(512)
void k_sparse_conv(const __bf16* __restrict__ in_bf, const __bf16* __restrict__ wfrag,
                   const int* __restrict__ tab, float* __restrict__ out, int M,
                   float* __restrict__ g_sum, float* __restrict__ g_sq) {
    __shared__ __bf16 sB[KV * 8 * 512];   // 216KB: all 27x8 B-fragments
    __shared__ float s_sum[64], s_sq[64];
    int tid = threadIdx.x;
    if (tid < 64) { s_sum[tid] = 0.f; s_sq[tid] = 0.f; }
    {   // cooperative staging: 13824 x uint4
        const uint4* src = (const uint4*)wfrag;
        uint4* dst = (uint4*)sB;
        for (int i = tid; i < KV * 8 * 512 * 2 / 16; i += 512) dst[i] = src[i];
    }
    __syncthreads();

    int lane = tid & 31;
    int wave = tid >> 5;
    long long rowbase = ((long long)blockIdx.x * 32 + wave * 2) * 16; // 512 rows/block
    bool active = rowbase < M;

    int r0 = (int)rowbase + (lane & 15);
    int r1 = r0 + 16;
    bool v0 = r0 < M, v1 = r1 < M;
    int base = (lane < 16) ? 0 : 8;

    v8f acc0[4] = {}, acc1[4] = {};
    if (active) {
#pragma unroll
        for (int k = 0; k < KV; k++) {
            int i0 = v0 ? tab[(long long)k * (M + 1) + r0] : M;
            int i1 = v1 ? tab[(long long)k * (M + 1) + r1] : M;
            if (k + 1 < KV)  // gfx1250 global_prefetch_b8 on next offset's index line
                __builtin_prefetch(&tab[(long long)(k + 1) * (M + 1) + r0], 0, 1);
            const __bf16* ar0 = in_bf + (long long)i0 * 64;
            const __bf16* ar1 = in_bf + (long long)i1 * 64;
            // batched A loads (global b128) ...
            v16bf a00 = ld16(ar0 + base,      ar0 + 16 + base);
            v16bf a01 = ld16(ar0 + 32 + base, ar0 + 48 + base);
            v16bf a10 = ld16(ar1 + base,      ar1 + 16 + base);
            v16bf a11 = ld16(ar1 + 32 + base, ar1 + 48 + base);
            // ... batched B loads (LDS ds_load_b128) ...
            const __bf16* bk = sB + k * 4096 + lane * 16;
            v16bf b[8];
#pragma unroll
            for (int f = 0; f < 8; f++)
                b[f] = ld16(bk + f * 512, bk + f * 512 + 8);
            // ... then one 16-WMMA burst
#pragma unroll
            for (int t = 0; t < 4; t++) {
                acc0[t] = WMMA_BF16(a00, b[t], acc0[t]);
                acc1[t] = WMMA_BF16(a10, b[t], acc1[t]);
            }
#pragma unroll
            for (int t = 0; t < 4; t++) {
                acc0[t] = WMMA_BF16(a01, b[4 + t], acc0[t]);
                acc1[t] = WMMA_BF16(a11, b[4 + t], acc1[t]);
            }
        }
        // epilogue: store + BN stats for both tiles
        int half = (lane < 16) ? 0 : 8;
#pragma unroll
        for (int t = 0; t < 4; t++) {
            int c = 16 * t + (lane & 15);
            float s = 0.f, q = 0.f;
#pragma unroll
            for (int rr = 0; rr < 8; rr++) {
                long long row = rowbase + half + rr;
                if (row < M) {
                    float x = acc0[t][rr];
                    out[row * 64 + c] = x;
                    s += x; q += x * x;
                }
            }
#pragma unroll
            for (int rr = 0; rr < 8; rr++) {
                long long row = rowbase + 16 + half + rr;
                if (row < M) {
                    float x = acc1[t][rr];
                    out[row * 64 + c] = x;
                    s += x; q += x * x;
                }
            }
            atomicAdd(&s_sum[c], s);
            atomicAdd(&s_sq[c], q);
        }
    }
    __syncthreads();
    if (tid < 64) {
        atomicAdd(&g_sum[tid], s_sum[tid]);
        atomicAdd(&g_sq[tid], s_sq[tid]);
    }
}

// ---------------------------------------------------------------------------
// mid apply: in_bf = bf16(ELU(BN(conv)))   (pad row M stays zero)
// ---------------------------------------------------------------------------
__global__ void k_apply_mid(const float* __restrict__ conv, const float* __restrict__ scale,
                            const float* __restrict__ bias, __bf16* __restrict__ in_bf,
                            long long n) {
    long long i = (long long)blockIdx.x * blockDim.x + threadIdx.x;
    if (i < n) {
        int c = (int)(i & 63);
        in_bf[i] = (__bf16)elu1(scale[c] * conv[i] + bias[c]);
    }
}

// ---------------------------------------------------------------------------
// final apply: out = ELU(ELU(BN(conv)) + down)
// ---------------------------------------------------------------------------
__global__ void k_apply_final(const float* __restrict__ conv, const float* __restrict__ scale,
                              const float* __restrict__ bias, const float* __restrict__ down_f,
                              float* __restrict__ out, long long n) {
    long long i = (long long)blockIdx.x * blockDim.x + threadIdx.x;
    if (i < n) {
        int c = (int)(i & 63);
        float o = elu1(scale[c] * conv[i] + bias[c]);
        out[i] = elu1(o + down_f[i]);
    }
}

// ---------------------------------------------------------------------------
// host orchestration
// ---------------------------------------------------------------------------
static inline unsigned ceil_div_ll(long long a, long long b) { return (unsigned)((a + b - 1) / b); }

extern "C" void kernel_launch(void* const* d_in, const int* in_sizes, int n_in,
                              void* d_out, int out_size, void* d_ws, size_t ws_size,
                              hipStream_t stream) {
    (void)n_in; (void)ws_size;
    const float* feats    = (const float*)d_in[0];
    const float* Wd       = (const float*)d_in[1];
    const float* g0       = (const float*)d_in[2];
    const float* b0       = (const float*)d_in[3];
    const float* W1       = (const float*)d_in[4];
    const float* g1       = (const float*)d_in[5];
    const float* b1       = (const float*)d_in[6];
    const float* W2       = (const float*)d_in[7];
    const float* g2       = (const float*)d_in[8];
    const float* b2       = (const float*)d_in[9];
    const int*   pool_seg = (const int*)d_in[10];
    const int*   conv_in  = (const int*)d_in[11];
    const int*   conv_out = (const int*)d_in[12];

    const int N = in_sizes[0] / 32;          // sparse input points
    const int L = in_sizes[11] / KV;         // padded pair-list length
    const int M = out_size / 64;             // pooled points (output rows)

    // --- workspace carve (256B aligned) ---
    char* p = (char*)d_ws;
    auto carve = [&](size_t bytes) -> char* {
        char* r = p; p += (bytes + 255) & ~(size_t)255; return r;
    };
    float*    st      = (float*)carve(3 * 256 * sizeof(float)); // 3 BN stages x {sum,sq,scale,bias}[64]
    __bf16*   wdfrag  = (__bf16*)carve((size_t)4 * 512 * 2);
    __bf16*   w1frag  = (__bf16*)carve((size_t)KV * 8 * 512 * 2);
    __bf16*   w2frag  = (__bf16*)carve((size_t)KV * 8 * 512 * 2);
    unsigned* down_u  = (unsigned*)carve((size_t)M * 64 * 4);   // pool max -> fp32 down
    __bf16*   in_bf   = (__bf16*)carve((size_t)(M + 1) * 64 * 2);
    float*    convb   = (float*)carve((size_t)M * 64 * 4);
    int*      tab     = (int*)carve((size_t)KV * (M + 1) * 4);

    float* sum0 = st;           float* sq0 = st + 64;   float* sc0 = st + 128;  float* bi0 = st + 192;
    float* sum1 = st + 256;     float* sq1 = st + 320;  float* sc1 = st + 384;  float* bi1 = st + 448;
    float* sum2 = st + 512;     float* sq2 = st + 576;  float* sc2 = st + 640;  float* bi2 = st + 704;

    const long long ME = (long long)M * 64;

    // --- init ---
    k_fill_u32<<<ceil_div_ll(768, 256), 256, 0, stream>>>((unsigned*)st, 768, 0u);
    k_fill_u32<<<ceil_div_ll(ME, 256), 256, 0, stream>>>(down_u, ME, 0u); // encoded -inf
    k_fill_u32<<<1, 32, 0, stream>>>((unsigned*)(in_bf + ME), 32, 0u);    // zero pad row
    k_fill_u32<<<ceil_div_ll((long long)KV * (M + 1), 256), 256, 0, stream>>>(
        (unsigned*)tab, (long long)KV * (M + 1), (unsigned)M);

    // --- weight fragment packing (bf16, WMMA B layout) ---
    k_build_bfrag<<<ceil_div_ll(2048, 256), 256, 0, stream>>>(Wd, wdfrag, 1, 1);
    k_build_bfrag<<<ceil_div_ll(KV * 2 * 4 * 512, 256), 256, 0, stream>>>(W1, w1frag, KV, 2);
    k_build_bfrag<<<ceil_div_ll(KV * 2 * 4 * 512, 256), 256, 0, stream>>>(W2, w2frag, KV, 2);

    // --- invert scatter lists into gather table ---
    k_build_idx<<<ceil_div_ll((long long)KV * L, 256), 256, 0, stream>>>(conv_in, conv_out, tab, L, M);

    // --- 1x1 conv: stats pass, finalize, apply+pool (GEMM recomputed: HBM-cheap) ---
    unsigned gblk = ceil_div_ll(((long long)N + 15) / 16, 8);
    k_down_gemm<0><<<gblk, 256, 0, stream>>>(feats, wdfrag, N, sum0, sq0,
                                             nullptr, nullptr, nullptr, nullptr);
    k_bn_finalize<<<1, 64, 0, stream>>>(g0, b0, sum0, sq0, (float)N, sc0, bi0);
    k_down_gemm<1><<<gblk, 256, 0, stream>>>(feats, wdfrag, N, nullptr, nullptr,
                                             sc0, bi0, pool_seg, down_u);
    k_decode_pool<<<ceil_div_ll(ME, 256), 256, 0, stream>>>(down_u, in_bf, ME);

    // --- LUConv block 1 ---
    unsigned cblk = ceil_div_ll(M, 512);     // 512 rows per block
    k_sparse_conv<<<cblk, 512, 0, stream>>>(in_bf, w1frag, tab, convb, M, sum1, sq1);
    k_bn_finalize<<<1, 64, 0, stream>>>(g1, b1, sum1, sq1, (float)M, sc1, bi1);
    k_apply_mid<<<ceil_div_ll(ME, 256), 256, 0, stream>>>(convb, sc1, bi1, in_bf, ME);

    // --- LUConv block 2 + residual ---
    k_sparse_conv<<<cblk, 512, 0, stream>>>(in_bf, w2frag, tab, convb, M, sum2, sq2);
    k_bn_finalize<<<1, 64, 0, stream>>>(g2, b2, sum2, sq2, (float)M, sc2, bi2);
    k_apply_final<<<ceil_div_ll(ME, 256), 256, 0, stream>>>(convb, sc2, bi2,
                                                            (const float*)down_u,
                                                            (float*)d_out, ME);
}
